// VanillaStellarModel_69999376990830
// MI455X (gfx1250) — matrix operand order repaired and checked by reading, not compiled, not run on hardware
//
#include <hip/hip_runtime.h>
#include <hip/hip_bf16.h>
#include <math.h>

typedef __attribute__((ext_vector_type(16))) __bf16 bf16x16;
typedef __attribute__((ext_vector_type(8)))  __bf16 bf16x8;
typedef __attribute__((ext_vector_type(4)))  __bf16 bf16x4;
typedef __attribute__((ext_vector_type(8)))  float  f32x8;
typedef __attribute__((ext_vector_type(4)))  unsigned int u32x4;
typedef __attribute__((ext_vector_type(8)))  int    i32x8;
typedef __attribute__((ext_vector_type(4)))  int    i32x4;

#define HID   128
#define INDIM 64
#define NCLS  50
#define NPAD  64   // classifier N padded to 4 WMMA tiles

// ---------------------------------------------------------------------------
// TDM: 1-D DMA of `bytes` (multiple of 8, <= 512KB) from global to LDS.
// D# per CDNA5 ISA (group0/group1; groups 2,3 zero).
// Issue from ONE wave (per-wave op, EXEC ignored), then s_wait_tensorcnt.
// This toolchain exposes the 6-arg builtin (g0, g1, g2, g3, extra, cpol).
// ---------------------------------------------------------------------------
__device__ __forceinline__ void tdm_load_to_lds(void* ldsDst, const void* src,
                                                unsigned bytes) {
  unsigned n8 = bytes >> 3;                       // elements of data_size=8B
  unsigned long long ga = (unsigned long long)(uintptr_t)src;
  u32x4 g0;
  g0[0] = 1u;                                     // count=1, user mode, no gather
  g0[1] = (unsigned)(uintptr_t)ldsDst;            // lds_addr (low 32 bits = LDS offset)
  g0[2] = (unsigned)ga;                           // global_addr[31:0]
  g0[3] = (unsigned)(((ga >> 32) & 0x01FFFFFFull) | (2u << 30)); // addr[56:32] | type=2
  i32x8 g1;
  g1[0] = (int)(3u << 16);                        // data_size = 8B; no flags
  g1[1] = (int)((n8 & 0xFFFFu) << 16);            // tensor_dim0[15:0]  @ bit48
  g1[2] = (int)((n8 >> 16) | (1u << 16));         // tensor_dim0[31:16], tensor_dim1=1
  g1[3] = (int)(n8 << 16);                        // tile_dim0 @ bit112
  g1[4] = 1;                                      // tile_dim1 = 1
  g1[5] = (int)n8;                                // tensor_dim0_stride lo32
  g1[6] = 0;
  g1[7] = 0;
  i32x4 z4 = {};
  i32x8 z8 = {};
  __builtin_amdgcn_tensor_load_to_lds(g0, g1, z4, z4, z8, 0);
}

// ---------------------------------------------------------------------------
// Fragment helpers (wave32 WMMA 16x16x32 bf16 layouts per CDNA5 ISA 7.12.2)
// A: lane l -> row (l&15); K elems: [kb+8*half .. +7] and [kb+16+8*half .. +7]
// B: lane l -> col (l&15); K elems: kb+8*half+(i&7)+16*(i>>3), i=0..15
// C/D: elem r -> row r+8*half, col (l&15)
// B is PRE-PACKED so each lane's 16 elems are 32 contiguous bytes.
// ---------------------------------------------------------------------------
__device__ __forceinline__ bf16x16 load_a_f32(const float* rowp, int kb, int half) {
  const float4* p = (const float4*)(rowp + kb + 8 * half);
  float4 c0 = p[0], c1 = p[1], c2 = p[4], c3 = p[5];   // +16 floats = +4 float4
  bf16x16 a;
  a[0]=(__bf16)c0.x;  a[1]=(__bf16)c0.y;  a[2]=(__bf16)c0.z;  a[3]=(__bf16)c0.w;
  a[4]=(__bf16)c1.x;  a[5]=(__bf16)c1.y;  a[6]=(__bf16)c1.z;  a[7]=(__bf16)c1.w;
  a[8]=(__bf16)c2.x;  a[9]=(__bf16)c2.y;  a[10]=(__bf16)c2.z; a[11]=(__bf16)c2.w;
  a[12]=(__bf16)c3.x; a[13]=(__bf16)c3.y; a[14]=(__bf16)c3.z; a[15]=(__bf16)c3.w;
  return a;
}

__device__ __forceinline__ bf16x16 load_a_f32_scaled(const float* rowp, int kb,
                                                     int half, float sc) {
  const float4* p = (const float4*)(rowp + kb + 8 * half);
  float4 c0 = p[0], c1 = p[1], c2 = p[4], c3 = p[5];
  bf16x16 a;
  a[0]=(__bf16)(c0.x*sc);  a[1]=(__bf16)(c0.y*sc);  a[2]=(__bf16)(c0.z*sc);  a[3]=(__bf16)(c0.w*sc);
  a[4]=(__bf16)(c1.x*sc);  a[5]=(__bf16)(c1.y*sc);  a[6]=(__bf16)(c1.z*sc);  a[7]=(__bf16)(c1.w*sc);
  a[8]=(__bf16)(c2.x*sc);  a[9]=(__bf16)(c2.y*sc);  a[10]=(__bf16)(c2.z*sc); a[11]=(__bf16)(c2.w*sc);
  a[12]=(__bf16)(c3.x*sc); a[13]=(__bf16)(c3.y*sc); a[14]=(__bf16)(c3.z*sc); a[15]=(__bf16)(c3.w*sc);
  return a;
}

__device__ __forceinline__ bf16x16 load_a_bf16(const __bf16* rowp, int kb, int half) {
  const bf16x8* p = (const bf16x8*)(rowp + kb + 8 * half);
  bf16x8 lo = p[0], hi = p[2];                          // +16 elems = +2 chunks
  return __builtin_shufflevector(lo, hi, 0,1,2,3,4,5,6,7,8,9,10,11,12,13,14,15);
}

// packed fragment: frag f, lane l -> 32B at ((f*32 + l) * 16) elements
__device__ __forceinline__ bf16x16 frag_ld(const __bf16* base, int f, int lane) {
  return *(const bf16x16*)(base + (((size_t)f * 32 + lane) << 4));
}

#define WMMA_BF16(A, B, C) \
  __builtin_amdgcn_wmma_f32_16x16x32_bf16(false, (A), false, (B), (short)0, (C), false, false)

// ---------------------------------------------------------------------------
// Prep kernels: convert + pack weights into WMMA B-fragment order.
// packed[( (t*ksteps+s)*32 + lane )*16 + i] = W[(32s + 8*(lane>>4) + (i&7) + 16*(i>>3))*ncols + 16t + (lane&15)]
// ---------------------------------------------------------------------------
__global__ void pack_w_f32(const float* __restrict__ W, __bf16* __restrict__ P,
                           int K, int ncols) {
  int idx = blockIdx.x * blockDim.x + threadIdx.x;
  if (idx >= K * ncols) return;
  int i = idx & 15, lane = (idx >> 4) & 31, f = idx >> 9;
  int ksteps = K >> 5;
  int s = f % ksteps, t = f / ksteps;
  int k = 32 * s + 8 * (lane >> 4) + (i & 7) + 16 * (i >> 3);
  int c = 16 * t + (lane & 15);
  P[idx] = (__bf16)W[(size_t)k * ncols + c];
}

__global__ void pack_w_bf16(const __bf16* __restrict__ W, __bf16* __restrict__ P,
                            int K, int ncols) {
  int idx = blockIdx.x * blockDim.x + threadIdx.x;
  if (idx >= K * ncols) return;
  int i = idx & 15, lane = (idx >> 4) & 31, f = idx >> 9;
  int ksteps = K >> 5;
  int s = f % ksteps, t = f / ksteps;
  int k = 32 * s + 8 * (lane >> 4) + (i & 7) + 16 * (i >> 3);
  int c = 16 * t + (lane & 15);
  P[idx] = W[(size_t)k * ncols + c];
}

// Normalize each of the 50 columns of W_cls [128,50] -> wn row-major [128,64] bf16
__global__ __launch_bounds__(128) void wcls_norm_kernel(const float* __restrict__ Wc,
                                                        __bf16* __restrict__ wn) {
  int c = blockIdx.x;    // 0..49
  int t = threadIdx.x;   // 0..127
  float v = Wc[(size_t)t * NCLS + c];
  __shared__ float red[128];
  red[t] = v * v;
  __syncthreads();
#pragma unroll
  for (int s = 64; s > 0; s >>= 1) {
    if (t < s) red[t] += red[t + s];
    __syncthreads();
  }
  float inv = 1.0f / fmaxf(sqrtf(red[0]), 1e-12f);
  wn[(size_t)t * NPAD + c] = (__bf16)(v * inv);
}

// ---------------------------------------------------------------------------
// Encoder: feat = relu(x @ W_in + b_in) -> bf16 [N,128]
// block = 64 rows; 8 waves = 8 col tiles; K=64 (2 steps); W_in staged via TDM
// ---------------------------------------------------------------------------
__global__ __launch_bounds__(256) void encoder_kernel(
    const float* __restrict__ x, const __bf16* __restrict__ WinP,
    const float* __restrict__ b_in, __bf16* __restrict__ featb, int n) {
  __shared__ __align__(32) __bf16 ldsW[INDIM * HID];   // 16 KB packed
  int tid = threadIdx.x, lane = tid & 31, wave = tid >> 5;
  if (tid < 32) {
    tdm_load_to_lds(ldsW, WinP, INDIM * HID * sizeof(__bf16));
    __builtin_amdgcn_s_wait_tensorcnt(0);
  }
  __syncthreads();

  int half = lane >> 4, q = lane & 15, col = wave * 16 + q;
  int row0 = blockIdx.x * 64;

  f32x8 acc[4] = {};
#pragma unroll
  for (int s = 0; s < 2; ++s) {
    bf16x16 b = frag_ld(ldsW, wave * 2 + s, lane);
#pragma unroll
    for (int rt = 0; rt < 4; ++rt) {
      int m = row0 + rt * 16 + q;
      if (m >= n) m = n - 1;
      bf16x16 a = load_a_f32(x + (size_t)m * INDIM, s * 32, half);
      acc[rt] = WMMA_BF16(a, b, acc[rt]);
    }
  }
  float bias = b_in[col];
  if (row0 + 64 <= n) {
#pragma unroll
    for (int rt = 0; rt < 4; ++rt)
#pragma unroll
      for (int r = 0; r < 8; ++r) {
        int row = row0 + rt * 16 + r + 8 * half;
        float v = acc[rt][r] + bias;
        featb[(size_t)row * HID + col] = (__bf16)(v > 0.f ? v : 0.f);
      }
  } else {
#pragma unroll
    for (int rt = 0; rt < 4; ++rt)
#pragma unroll
      for (int r = 0; r < 8; ++r) {
        int row = row0 + rt * 16 + r + 8 * half;
        if (row < n) {
          float v = acc[rt][r] + bias;
          featb[(size_t)row * HID + col] = (__bf16)(v > 0.f ? v : 0.f);
        }
      }
  }
}

// ---------------------------------------------------------------------------
// Edge scatter: agg[dst] += feat[src]; cnt[dst] += 1.  One wave per edge.
// ---------------------------------------------------------------------------
__global__ __launch_bounds__(256) void edge_scatter_kernel(
    const long long* __restrict__ ei, const __bf16* __restrict__ featb,
    float* __restrict__ agg, float* __restrict__ cnt, int E) {
  int e = blockIdx.x * 8 + (threadIdx.x >> 5);
  if (e >= E) return;
  int lane = threadIdx.x & 31;
  int src = (int)ei[(size_t)e];
  int dst = (int)ei[(size_t)E + e];

  bf16x4 v = *(const bf16x4*)(featb + (size_t)src * HID + lane * 4);
  float* ap = agg + (size_t)dst * HID + lane * 4;
#pragma unroll
  for (int i = 0; i < 4; ++i) atomicAdd(&ap[i], (float)v[i]);
  if (lane == 0) atomicAdd(&cnt[dst], 1.0f);
}

// ---------------------------------------------------------------------------
// Combine: out_feat = (agg/max(cnt,1)) @ W_l + b_l + feat @ W_r
// block = 64 rows; 8 waves = 8 col tiles; K=128 (4 steps); W_l|W_r staged via TDM
// ---------------------------------------------------------------------------
__global__ __launch_bounds__(256) void combine_kernel(
    const float* __restrict__ agg, const float* __restrict__ cnt,
    const __bf16* __restrict__ featb, const __bf16* __restrict__ WlrP,
    const float* __restrict__ b_l, float* __restrict__ out_feat,
    __bf16* __restrict__ ofb, int n) {
  __shared__ __align__(32) __bf16 ldsW[2 * HID * HID];  // 64 KB: [Wl | Wr] packed
  int tid = threadIdx.x, lane = tid & 31, wave = tid >> 5;
  if (tid < 32) {
    tdm_load_to_lds(ldsW, WlrP, 2 * HID * HID * sizeof(__bf16));
    __builtin_amdgcn_s_wait_tensorcnt(0);
  }
  __syncthreads();

  int half = lane >> 4, q = lane & 15, col = wave * 16 + q;
  int row0 = blockIdx.x * 64;

  const float*  ar[4];
  const __bf16* fr[4];
  float inv[4];
#pragma unroll
  for (int rt = 0; rt < 4; ++rt) {
    int m = row0 + rt * 16 + q;
    if (m >= n) m = n - 1;
    ar[rt]  = agg   + (size_t)m * HID;
    fr[rt]  = featb + (size_t)m * HID;
    inv[rt] = 1.0f / fmaxf(cnt[m], 1.0f);
  }

  f32x8 acc[4] = {};
#pragma unroll
  for (int s = 0; s < 4; ++s) {
    bf16x16 bl = frag_ld(ldsW,             wave * 4 + s, lane);
    bf16x16 br = frag_ld(ldsW + HID * HID, wave * 4 + s, lane);
#pragma unroll
    for (int rt = 0; rt < 4; ++rt) {
      bf16x16 a1 = load_a_f32_scaled(ar[rt], s * 32, half, inv[rt]);
      acc[rt] = WMMA_BF16(a1, bl, acc[rt]);
      bf16x16 a2 = load_a_bf16(fr[rt], s * 32, half);
      acc[rt] = WMMA_BF16(a2, br, acc[rt]);
    }
  }
  float bias = b_l[col];
  if (row0 + 64 <= n) {
#pragma unroll
    for (int rt = 0; rt < 4; ++rt)
#pragma unroll
      for (int r = 0; r < 8; ++r) {
        int row = row0 + rt * 16 + r + 8 * half;
        float v = acc[rt][r] + bias;
        out_feat[(size_t)row * HID + col] = v;
        ofb[(size_t)row * HID + col]      = (__bf16)v;
      }
  } else {
#pragma unroll
    for (int rt = 0; rt < 4; ++rt)
#pragma unroll
      for (int r = 0; r < 8; ++r) {
        int row = row0 + rt * 16 + r + 8 * half;
        if (row < n) {
          float v = acc[rt][r] + bias;
          out_feat[(size_t)row * HID + col] = v;
          ofb[(size_t)row * HID + col]      = (__bf16)v;
        }
      }
  }
}

// ---------------------------------------------------------------------------
// Classifier: out = (10/max(||of_i||,eps)) * (of_i @ wn)   [N,50]
// block = 64 rows; 4 waves = 4 col tiles (64 padded); K=128; wn staged via TDM
// ---------------------------------------------------------------------------
__global__ __launch_bounds__(128) void cls_kernel(
    const float* __restrict__ out_feat, const __bf16* __restrict__ ofb,
    const __bf16* __restrict__ wnP, float* __restrict__ out, int n) {
  __shared__ __align__(32) __bf16 ldsW[HID * NPAD];    // 16 KB packed
  __shared__ float part[128];
  __shared__ float scale[64];
  int tid = threadIdx.x, lane = tid & 31, wave = tid >> 5;
  if (tid < 32) {
    tdm_load_to_lds(ldsW, wnP, HID * NPAD * sizeof(__bf16));
    __builtin_amdgcn_s_wait_tensorcnt(0);
  }

  int row0 = blockIdx.x * 64;
  // per-row 2-norm of out_feat (f32): 2 threads per row, 64 elems each
  {
    int r = tid >> 1, hc = tid & 1;
    int row = row0 + r; if (row >= n) row = n - 1;
    const float4* orow = (const float4*)(out_feat + (size_t)row * HID + hc * 64);
    float s = 0.f;
#pragma unroll
    for (int i = 0; i < 16; ++i) {
      float4 v = orow[i];
      s += v.x * v.x + v.y * v.y + v.z * v.z + v.w * v.w;
    }
    part[tid] = s;
  }
  __syncthreads();
  if ((tid & 1) == 0) {
    int r = tid >> 1;
    scale[r] = 10.0f / fmaxf(sqrtf(part[2 * r] + part[2 * r + 1]), 1e-12f);
  }
  __syncthreads();

  int half = lane >> 4, q = lane & 15, col = wave * 16 + q;
  const __bf16* fr[4];
#pragma unroll
  for (int rt = 0; rt < 4; ++rt) {
    int m = row0 + rt * 16 + q;
    if (m >= n) m = n - 1;
    fr[rt] = ofb + (size_t)m * HID;
  }

  f32x8 acc[4] = {};
#pragma unroll
  for (int s = 0; s < 4; ++s) {
    bf16x16 b = frag_ld(ldsW, wave * 4 + s, lane);
#pragma unroll
    for (int rt = 0; rt < 4; ++rt) {
      bf16x16 a = load_a_bf16(fr[rt], s * 32, half);
      acc[rt] = WMMA_BF16(a, b, acc[rt]);
    }
  }
  if (col < NCLS) {
    bool full = (row0 + 64 <= n);
#pragma unroll
    for (int rt = 0; rt < 4; ++rt)
#pragma unroll
      for (int r = 0; r < 8; ++r) {
        int row = row0 + rt * 16 + r + 8 * half;
        if (full || row < n)
          out[(size_t)row * NCLS + col] = acc[rt][r] * scale[rt * 16 + r + 8 * half];
      }
  }
}

// ---------------------------------------------------------------------------
extern "C" void kernel_launch(void* const* d_in, const int* in_sizes, int n_in,
                              void* d_out, int out_size, void* d_ws, size_t ws_size,
                              hipStream_t stream) {
  const float*     x     = (const float*)d_in[0];
  const long long* ei    = (const long long*)d_in[1];
  const float*     W_in  = (const float*)d_in[2];
  const float*     b_in  = (const float*)d_in[3];
  const float*     W_l   = (const float*)d_in[4];
  const float*     b_l   = (const float*)d_in[5];
  const float*     W_r   = (const float*)d_in[6];
  const float*     W_cls = (const float*)d_in[7];

  int n = in_sizes[0] / INDIM;     // 100000
  int E = in_sizes[1] / 2;         // 1600000

  char*  ws  = (char*)d_ws;
  size_t off = 0;
  auto take = [&](size_t bytes) -> void* {
    void* p = ws + off;
    off += (bytes + 255) & ~(size_t)255;
    return p;
  };
  __bf16* featb  = (__bf16*)take((size_t)n * HID * sizeof(__bf16));
  float*  agg    = (float*) take((size_t)n * HID * sizeof(float));
  float*  cnt    = (float*) take((size_t)n * sizeof(float));
  __bf16* ofb    = (__bf16*)take((size_t)n * HID * sizeof(__bf16));
  __bf16* WinP   = (__bf16*)take((size_t)INDIM * HID * sizeof(__bf16));
  __bf16* WlrP   = (__bf16*)take((size_t)2 * HID * HID * sizeof(__bf16)); // [Wl|Wr]
  __bf16* wn_rm  = (__bf16*)take((size_t)HID * NPAD * sizeof(__bf16));
  __bf16* wnP    = (__bf16*)take((size_t)HID * NPAD * sizeof(__bf16));

  float* out      = (float*)d_out;              // [N, 50]
  float* out_feat = out + (size_t)n * NCLS;     // [N, 128]

  (void)hipMemsetAsync(agg,   0, (size_t)n * HID * sizeof(float), stream);
  (void)hipMemsetAsync(cnt,   0, (size_t)n * sizeof(float), stream);
  (void)hipMemsetAsync(wn_rm, 0, (size_t)HID * NPAD * sizeof(__bf16), stream);

  // weight prep: convert + pack into WMMA fragment order
  pack_w_f32<<<(INDIM * HID + 255) / 256, 256, 0, stream>>>(W_in, WinP, INDIM, HID);
  pack_w_f32<<<(HID * HID + 255) / 256, 256, 0, stream>>>(W_l, WlrP, HID, HID);
  pack_w_f32<<<(HID * HID + 255) / 256, 256, 0, stream>>>(W_r, WlrP + HID * HID, HID, HID);
  wcls_norm_kernel<<<NCLS, 128, 0, stream>>>(W_cls, wn_rm);
  pack_w_bf16<<<(HID * NPAD + 255) / 256, 256, 0, stream>>>(wn_rm, wnP, HID, NPAD);

  int rblocks = (n + 63) / 64;
  encoder_kernel<<<rblocks, 256, 0, stream>>>(x, WinP, b_in, featb, n);
  edge_scatter_kernel<<<(E + 7) / 8, 256, 0, stream>>>(ei, featb, agg, cnt, E);
  combine_kernel<<<rblocks, 256, 0, stream>>>(agg, cnt, featb, WlrP, b_l,
                                              out_feat, ofb, n);
  cls_kernel<<<rblocks, 128, 0, stream>>>(out_feat, ofb, wnP, out, n);
}
